// HetAgg_77738908058621
// MI455X (gfx1250) — matrix operand order repaired and checked by reading, not compiled, not run on hardware
//
#include <hip/hip_runtime.h>

typedef _Float16 half4  __attribute__((ext_vector_type(4)));
typedef _Float16 half8  __attribute__((ext_vector_type(8)));
typedef _Float16 half16 __attribute__((ext_vector_type(16)));
typedef float    v8f    __attribute__((ext_vector_type(8)));

#define D    128     // embed dim
#define SS   10      // neighbor steps
#define LL   2       // layers
#define TT   3       // node types
#define ROWS 16      // batch rows per block (one wave handles a 16-row M tile)

// ---- fast branchless transcendentals (avoid libm's EXEC-divergent paths) ---
__device__ __forceinline__ float fast_tanh(float x) {
#if __has_builtin(__builtin_amdgcn_tanhf)
  return __builtin_amdgcn_tanhf(x);            // hw V_TANH_F32 on gfx1250
#else
  const float ax = __builtin_fabsf(x);
  const float e  = __builtin_amdgcn_exp2f(ax * -2.885390081777927f); // -2*log2(e)
  const float t  = (1.0f - e) * __builtin_amdgcn_rcpf(1.0f + e);
  return __builtin_copysignf(t, x);
#endif
}

__device__ __forceinline__ float fast_exp(float x) {
  return __builtin_amdgcn_exp2f(x * 1.4426950408889634f);
}

// ---- WMMA fragment loads (wave32, 16x16x32 f16) -------------------------
// A (16x32, MxK): lanes 0-15 hold row M=lane, K = kt*32 + {0..7,16..23};
//                 lanes 16-31 hold row M=lane-16, K = kt*32 + {8..15,24..31}.
__device__ __forceinline__ half16 ldfragA(const _Float16* base, int kt, int lane) {
  const int row = lane & 15;
  const int kb  = kt * 32 + ((lane >> 4) << 3);
  const half8 lo = *reinterpret_cast<const half8*>(base + row * D + kb);
  const half8 hi = *reinterpret_cast<const half8*>(base + row * D + kb + 16);
  half16 f;
#pragma unroll
  for (int i = 0; i < 8; ++i) { f[i] = lo[i]; f[i + 8] = hi[i]; }
  return f;
}

// B (32x16, KxN), stored as W[n][k] row-major (so B[k][n] = W[n][k]):
// lanes 0-15 hold col N=lane with K = kt*32 + {0..7,16..23};
// lanes 16-31 hold col N=lane-16 with K = kt*32 + {8..15,24..31}.
__device__ __forceinline__ half16 ldfragB(const _Float16* wbase, int kt, int lane) {
  const int col = lane & 15;
  const int kb  = kt * 32 + ((lane >> 4) << 3);
  const half8 a = *reinterpret_cast<const half8*>(wbase + col * D + kb);
  const half8 b = *reinterpret_cast<const half8*>(wbase + col * D + kb + 16);
  half16 f;
#pragma unroll
  for (int i = 0; i < 8; ++i) { f[i] = a[i]; f[i + 8] = b[i]; }
  return f;
}

__device__ __forceinline__ v8f wmma_f16(half16 a, half16 b, v8f c) {
  return __builtin_amdgcn_wmma_f32_16x16x32_f16(false, a, false, b, (short)0, c,
                                                false, false);
}

__global__ __launch_bounds__(32) void hetagg_wmma_kernel(
    const int* __restrict__ c_ids, const int* __restrict__ pos_ids,
    const int* __restrict__ neg_ids,
    const int* __restrict__ n00, const int* __restrict__ n01, const int* __restrict__ n02,
    const int* __restrict__ n10, const int* __restrict__ n11, const int* __restrict__ n12,
    const int* __restrict__ n20, const int* __restrict__ n21, const int* __restrict__ n22,
    const float* __restrict__ emb0, const float* __restrict__ emb1,
    const float* __restrict__ emb2,
    const float* __restrict__ Wih, const float* __restrict__ Whh,
    const float* __restrict__ bih, const float* __restrict__ bhh,
    const float* __restrict__ attW, float* __restrict__ out) {
  // LDS (~146 KB): weights f16, gathered x f16, double-buffered h f16,
  // cur/agg f32, small attention scratch.
  __shared__ _Float16 s_wih[D * D];
  __shared__ _Float16 s_whh[D * D];
  __shared__ float    s_bias[D];
  __shared__ _Float16 s_x[SS * ROWS * D];
  __shared__ _Float16 s_h[2 * ROWS * D];
  __shared__ float    s_cur[ROWS * D];
  __shared__ float    s_agg[TT * ROWS * D];
  __shared__ int      s_ids[ROWS];
  __shared__ int      s_nids[ROWS * SS];
  __shared__ float    s_cs[ROWS];
  __shared__ float    s_sc[ROWS * 4];
  __shared__ float    s_sw[ROWS * 4];

  const int lane = threadIdx.x;
  const int tile = blockIdx.x;        // 768 tiles of 16 rows = 12288 rows
  const int run  = tile >> 8;         // 256 tiles per 4096-row run
  const int ntype = (run == 0) ? 0 : 1;
  const int* idlist = (run == 0) ? c_ids : (run == 1) ? pos_ids : neg_ids;

  // neighbor tables for this center type, and per-type embeddings
  const int* nghA = (ntype == 0) ? n00 : n10;
  const int* nghB = (ntype == 0) ? n01 : n11;
  const int* nghC = (ntype == 0) ? n02 : n12;
  const float* ecur = (ntype == 0) ? emb0 : emb1;

  if (lane < ROWS) s_ids[lane] = idlist[(tile & 255) * ROWS + lane];
  __syncthreads();

  // stage cur = embs[ntype][ids]  (float4 gathers)
  for (int i = lane; i < ROWS * D / 4; i += 32) {
    int r = i >> 5;
    float4 v = reinterpret_cast<const float4*>(ecur + (size_t)s_ids[r] * D)[i & 31];
    reinterpret_cast<float4*>(s_cur)[i] = v;
  }
  __syncthreads();

#pragma unroll 1
  for (int l = 0; l < LL; ++l) {
#pragma unroll 1
    for (int nt = 0; nt < TT; ++nt) {
      const int wsel = l * TT + nt;
      const float* wi = Wih + (size_t)wsel * D * D;
      const float* wh = Whh + (size_t)wsel * D * D;
      // stage weights f32 -> f16 in LDS
      for (int i = lane; i < D * D / 4; i += 32) {
        float4 a = reinterpret_cast<const float4*>(wi)[i];
        float4 b = reinterpret_cast<const float4*>(wh)[i];
        half4 ha = {(_Float16)a.x, (_Float16)a.y, (_Float16)a.z, (_Float16)a.w};
        half4 hb = {(_Float16)b.x, (_Float16)b.y, (_Float16)b.z, (_Float16)b.w};
        reinterpret_cast<half4*>(s_wih)[i] = ha;
        reinterpret_cast<half4*>(s_whh)[i] = hb;
      }
      for (int k = lane; k < D; k += 32)
        s_bias[k] = bih[wsel * D + k] + bhh[wsel * D + k];
      const int* ng = (nt == 0) ? nghA : (nt == 1) ? nghB : nghC;
      for (int i = lane; i < ROWS * SS; i += 32)
        s_nids[i] = ng[(size_t)s_ids[i / SS] * SS + (i % SS)];
      __syncthreads();

      // gather x[s][r][c] = embs[nt][nid][c] as f16
      const float* en = (nt == 0) ? emb0 : (nt == 1) ? emb1 : emb2;
      for (int i = lane; i < ROWS * SS * D / 4; i += 32) {
        int c4 = i & 31, rs = i >> 5;
        int r = rs / SS, s = rs % SS;
        float4 v = reinterpret_cast<const float4*>(en + (size_t)s_nids[rs] * D)[c4];
        half4 h = {(_Float16)v.x, (_Float16)v.y, (_Float16)v.z, (_Float16)v.w};
        reinterpret_cast<half4*>(s_x + (size_t)(s * ROWS + r) * D)[c4] = h;
      }
      for (int i = lane; i < ROWS * D; i += 32) s_h[i] = (_Float16)0.f;
      __syncthreads();

      // prefetch next stage's weights into L2/L0 while we crunch the RNN
      if (wsel + 1 < LL * TT) {
        const float* win = Wih + (size_t)(wsel + 1) * D * D;
        const float* whn = Whh + (size_t)(wsel + 1) * D * D;
#pragma unroll
        for (int i = 0; i < 16; ++i) {
          __builtin_prefetch(win + (size_t)i * 1024 + lane * 32, 0, 0);
          __builtin_prefetch(whn + (size_t)i * 1024 + lane * 32, 0, 0);
        }
      }

      v8f mean[8];
#pragma unroll
      for (int n = 0; n < 8; ++n) mean[n] = (v8f){0, 0, 0, 0, 0, 0, 0, 0};

      const int col16 = lane & 15;
      const int mro   = (lane >> 4) * 8;

#pragma unroll 1
      for (int t = 0; t < SS; ++t) {
        const _Float16* hp = s_h + (t & 1) * (ROWS * D);
        _Float16*       hn = s_h + ((t + 1) & 1) * (ROWS * D);
        half16 ax[4], ah[4];
#pragma unroll
        for (int kt = 0; kt < 4; ++kt) {
          ax[kt] = ldfragA(s_x + t * ROWS * D, kt, lane);
          ah[kt] = ldfragA(hp, kt, lane);
        }
#pragma unroll
        for (int ntile = 0; ntile < 8; ++ntile) {
          const float bv = s_bias[ntile * 16 + col16];
          v8f acc = {bv, bv, bv, bv, bv, bv, bv, bv};
#pragma unroll
          for (int kt = 0; kt < 4; ++kt)
            acc = wmma_f16(ax[kt], ldfragB(s_wih + ntile * 16 * D, kt, lane), acc);
#pragma unroll
          for (int kt = 0; kt < 4; ++kt)
            acc = wmma_f16(ah[kt], ldfragB(s_whh + ntile * 16 * D, kt, lane), acc);
          const int col = ntile * 16 + col16;
#pragma unroll
          for (int v = 0; v < 8; ++v) {
            float hv = fast_tanh(acc[v]);
            mean[ntile][v] += hv;
            hn[(v + mro) * D + col] = (_Float16)hv;
          }
        }
        __syncthreads();
      }
      // agg[nt] = mean of hidden states
#pragma unroll
      for (int ntile = 0; ntile < 8; ++ntile) {
        const int col = ntile * 16 + col16;
#pragma unroll
        for (int v = 0; v < 8; ++v)
          s_agg[nt * ROWS * D + (v + mro) * D + col] = mean[ntile][v] * (1.0f / SS);
      }
      __syncthreads();
    }  // nt

    // ---- attention over {cur, agg0, agg1, agg2} ----
    const float* aw = attW + (size_t)(l * TT + ntype) * (2 * D);
    if (lane < ROWS) {
      float s = 0.f;
      for (int c = 0; c < D; ++c) s += aw[c] * s_cur[lane * D + c];
      s_cs[lane] = s;
    }
    __syncthreads();
    for (int i = lane; i < ROWS * 4; i += 32) {
      int r = i >> 2, k = i & 3;
      const float* src = (k == 0) ? (s_cur + r * D) : (s_agg + (k - 1) * ROWS * D + r * D);
      float s = 0.f;
      for (int c = 0; c < D; ++c) s += aw[D + c] * src[c];
      s_sc[i] = s + s_cs[r];
    }
    __syncthreads();
    if (lane < ROWS) {
      float m = s_sc[lane * 4];
      for (int k = 1; k < 4; ++k) m = fmaxf(m, s_sc[lane * 4 + k]);
      float e[4], sum = 0.f;
      for (int k = 0; k < 4; ++k) { e[k] = fast_exp(s_sc[lane * 4 + k] - m); sum += e[k]; }
      float inv = 1.0f / sum;
      for (int k = 0; k < 4; ++k) s_sw[lane * 4 + k] = e[k] * inv;
    }
    __syncthreads();
    for (int i = lane; i < ROWS * D; i += 32) {
      int r = i >> 7;
      float v = s_sw[r * 4 + 0] * s_cur[i] + s_sw[r * 4 + 1] * s_agg[i] +
                s_sw[r * 4 + 2] * s_agg[ROWS * D + i] +
                s_sw[r * 4 + 3] * s_agg[2 * ROWS * D + i];
      s_cur[i] = (v > 0.f) ? v : 0.01f * v;
    }
    __syncthreads();
  }  // l

  // write out: rows are already laid out as [c | pos | neg] * 4096 * 128
  for (int i = lane; i < ROWS * D / 4; i += 32)
    reinterpret_cast<float4*>(out + (size_t)tile * ROWS * D)[i] =
        reinterpret_cast<const float4*>(s_cur)[i];
}

extern "C" void kernel_launch(void* const* d_in, const int* in_sizes, int n_in,
                              void* d_out, int out_size, void* d_ws, size_t ws_size,
                              hipStream_t stream) {
  (void)in_sizes; (void)n_in; (void)out_size; (void)d_ws; (void)ws_size;
  hetagg_wmma_kernel<<<768, 32, 0, stream>>>(
      (const int*)d_in[0], (const int*)d_in[1], (const int*)d_in[2],
      (const int*)d_in[3], (const int*)d_in[4], (const int*)d_in[5],
      (const int*)d_in[6], (const int*)d_in[7], (const int*)d_in[8],
      (const int*)d_in[9], (const int*)d_in[10], (const int*)d_in[11],
      (const float*)d_in[12], (const float*)d_in[13], (const float*)d_in[14],
      (const float*)d_in[15], (const float*)d_in[16],
      (const float*)d_in[17], (const float*)d_in[18],
      (const float*)d_in[19],
      (float*)d_out);
}
